// BiLSTM_CRF_74088185856710
// MI455X (gfx1250) — matrix operand order repaired and checked by reading, not compiled, not run on hardware
//
#include <hip/hip_runtime.h>
#include <hip/hip_bf16.h>

#define Bn 512
#define Tn 512
#define Kn 128
#define START_TAG 126
#define STOP_TAG 127

typedef __attribute__((ext_vector_type(16))) _Float16 v16h;
typedef __attribute__((ext_vector_type(8)))  _Float16 v8h;
typedef __attribute__((ext_vector_type(8)))  float    v8f;
typedef __attribute__((ext_vector_type(4)))  unsigned int u32x4;
typedef __attribute__((ext_vector_type(8)))  int      i32x8;
typedef __attribute__((ext_vector_type(4)))  int      i32x4;

#define USE_TDM (__has_builtin(__builtin_amdgcn_tensor_load_to_lds))

#if USE_TDM
// 2D tile load: 16 rows x 128 f32, row pitch T*K floats, into LDS at lds_off.
__device__ __forceinline__ void tdm_load_tile(const float* gbase, unsigned lds_off) {
    unsigned long long ga = (unsigned long long)(uintptr_t)gbase;
    u32x4 g0;
    g0.x = 1u;                                           // count=1, user descriptor
    g0.y = lds_off;                                      // LDS byte address
    g0.z = (unsigned)(ga & 0xFFFFFFFFu);                 // global_addr[31:0]
    g0.w = (unsigned)((ga >> 32) & 0x01FFFFFFu) | (2u << 30); // addr[56:32] | type=2
    i32x8 g1;
    g1[0] = (int)(2u << 16);        // data_size = 4B (code 2); workgroup_mask=0
    g1[1] = (int)(128u << 16);      // tensor_dim0[15:0] = 128 (bits 63:48)
    g1[2] = (int)(16u << 16);       // tensor_dim0 hi = 0 ; tensor_dim1[15:0] = 16
    g1[3] = (int)(128u << 16);      // tensor_dim1 hi = 0 ; tile_dim0 = 128
    g1[4] = 16;                     // tile_dim1 = 16 ; tile_dim2 = 0
    g1[5] = (int)(Tn * Kn);         // tensor_dim0_stride[31:0] = 65536 elements
    g1[6] = 0;
    g1[7] = 0;
    i32x4 gz4 = {0, 0, 0, 0};       // groups 2/3 unused (2D tensor)
    i32x8 gz8 = {0, 0, 0, 0, 0, 0, 0, 0};
    __builtin_amdgcn_tensor_load_to_lds(g0, g1, gz4, gz4, gz8, 0);
}
#endif

// ---------------------------------------------------------------------------
// Forward (log-partition) kernel: grid 32 x 256 threads, 16 batch rows/block.
// ---------------------------------------------------------------------------
__global__ __launch_bounds__(256) void crf_forward_kernel(
    const float* __restrict__ feats, const float* __restrict__ trans,
    float* __restrict__ fwd_out) {
    __shared__ __align__(16) _Float16 Ash[16 * 136];       // exp(alphaHat) f16
    __shared__ float lBuf[16 * 136];                       // pre-max alpha deltas
    __shared__ float featBuf[2][16 * Kn];                  // double-buffered feats tile
    __shared__ float partialMax[16 * 16];
    __shared__ float rowmaxArr[16];
    __shared__ float offArr[16];
    __shared__ float rmaxN[Kn];                            // rowmax of trans[n][:]
    __shared__ float tstop[Kn];                            // trans[STOP][n]

    const int tid  = threadIdx.x;
    const int wave = tid >> 5;       // 0..7 -> owns next-tags [16w, 16w+16)
    const int lane = tid & 31;
    const int nl   = lane & 15;      // column within 16-wide N tile
    const int hh   = lane >> 4;      // lane half (WMMA layout)
    const int b0   = blockIdx.x * 16;

    // ---- setup: transition row maxes, stop row, alpha0, offsets ----
    if (tid < Kn) {
        const float* tr = trans + tid * Kn;
        float mx = tr[0];
        for (int j = 1; j < Kn; ++j) mx = fmaxf(mx, tr[j]);
        rmaxN[tid] = mx;
        tstop[tid] = trans[STOP_TAG * Kn + tid];
    }
    if (tid < 16) offArr[tid] = 0.0f;
    {
        int m = tid >> 4, seg = tid & 15;
        #pragma unroll
        for (int j = 0; j < 8; ++j) {
            int k = seg * 8 + j;
            Ash[m * 136 + k] = (k == START_TAG) ? (_Float16)1.0f : (_Float16)0.0f;
        }
    }
    __syncthreads();

    // ---- B operand: exp(trans[n][k]-rmax[n]) resident in VGPRs for all T ----
    const int ncol = 16 * wave + nl;
    v16h breg[4];
    {
        const float rmN = rmaxN[ncol];
        const float* trow = trans + ncol * Kn;
        #pragma unroll
        for (int c = 0; c < 4; ++c) {
            v16h b;
            #pragma unroll
            for (int j = 0; j < 16; ++j) {
                int k = 32 * c + ((j < 8) ? 8 * hh + j : 16 + 8 * hh + (j - 8)); // B 32x16 f16 lane layout
                b[j] = (_Float16)__expf(trow[k] - rmN);
            }
            breg[c] = b;
        }
    }

#if USE_TDM
    // preload feats tile for t = 0
    if (wave == 0) {
        tdm_load_tile(feats + (size_t)b0 * Tn * Kn,
                      (unsigned)(uintptr_t)(void*)&featBuf[0][0]);
        __builtin_amdgcn_s_wait_tensorcnt(0);
    }
#endif
    __syncthreads();

    // ---- time scan ----
    for (int t = 0; t < Tn; ++t) {
#if USE_TDM
        if (wave == 0 && (t + 1) < Tn) {
            tdm_load_tile(feats + ((size_t)b0 * Tn + (t + 1)) * Kn,
                          (unsigned)(uintptr_t)(void*)&featBuf[(t + 1) & 1][0]);
        }
#endif
        // S[m][n] = sum_k exp(alphaHat[m][k]) * exp(trans[n][k]-rmax[n])  (WMMA)
        v8f acc = {};
        #pragma unroll
        for (int c = 0; c < 4; ++c) {
            const _Float16* ar = Ash + (lane & 15) * 136 + 32 * c + 8 * hh;
            v8h lo = *(const v8h*)(ar);
            v8h hi = *(const v8h*)(ar + 16);
            v16h a = __builtin_shufflevector(lo, hi, 0, 1, 2, 3, 4, 5, 6, 7,
                                             8, 9, 10, 11, 12, 13, 14, 15);
            acc = __builtin_amdgcn_wmma_f32_16x16x32_f16(
                false, a, false, breg[c], (short)0, acc, false, false);
        }
        // l[m][n] = rmax[n] + log S + feat[m][t][n]
        {
            const float rn = rmaxN[ncol];
#if USE_TDM
            const float* fb = &featBuf[t & 1][0];
#endif
            #pragma unroll
            for (int r = 0; r < 8; ++r) {
                int m = 8 * hh + r;
#if USE_TDM
                float fv = fb[m * Kn + ncol];
#else
                float fv = feats[((size_t)(b0 + m) * Tn + t) * Kn + ncol];
#endif
                float S = acc[r];
                lBuf[m * 136 + ncol] = rn + __logf(fmaxf(S, 1e-30f)) + fv;
            }
        }
        __syncthreads();
        // per-batch-row max of l (stabilizer / running offset)
        {
            int m = tid >> 4, seg = tid & 15;
            float mx = -3.0e38f;
            #pragma unroll
            for (int j = 0; j < 8; ++j)
                mx = fmaxf(mx, lBuf[m * 136 + seg * 8 + j]);
            partialMax[m * 16 + seg] = mx;
        }
        __syncthreads();
        if (tid < 16) {
            float r = partialMax[tid * 16];
            #pragma unroll
            for (int j = 1; j < 16; ++j) r = fmaxf(r, partialMax[tid * 16 + j]);
            rowmaxArr[tid] = r;
            offArr[tid] += r;
        }
        __syncthreads();
        // A_new = exp(l - rowmax) as f16 for next step's WMMA
        {
            int m = tid >> 4, seg = tid & 15;
            float rm = rowmaxArr[m];
            #pragma unroll
            for (int j = 0; j < 8; ++j) {
                int nn = seg * 8 + j;
                Ash[m * 136 + nn] = (_Float16)__expf(lBuf[m * 136 + nn] - rm);
            }
        }
#if USE_TDM
        if (wave == 0) __builtin_amdgcn_s_wait_tensorcnt(0);
#endif
        __syncthreads();
    }

    // forward[b] = off + log sum_n exp(alphaHatFinal[n] + trans[STOP][n])
    if (tid < 16) {
        float rm = rowmaxArr[tid];
        float s = 0.0f;
        for (int n = 0; n < Kn; ++n)
            s += __expf(lBuf[tid * 136 + n] - rm + tstop[n]);
        fwd_out[b0 + tid] = offArr[tid] + __logf(s);
    }
}

// ---------------------------------------------------------------------------
// Gold-path score: one thread per batch row.
// ---------------------------------------------------------------------------
__global__ __launch_bounds__(256) void crf_gold_kernel(
    const float* __restrict__ feats, const int* __restrict__ tags,
    const float* __restrict__ trans, float* __restrict__ gold_out) {
    int b = blockIdx.x * blockDim.x + threadIdx.x;
    if (b >= Bn) return;
    const int* tg = tags + (size_t)b * Tn;
    const float* fb = feats + (size_t)b * Tn * Kn;
    int prev = START_TAG;
    float s = 0.0f;
    for (int t = 0; t < Tn; ++t) {
        int cur = tg[t];
        s += trans[cur * Kn + prev] + fb[(size_t)t * Kn + cur];
        prev = cur;
    }
    s += trans[STOP_TAG * Kn + prev];
    gold_out[b] = s;
}

// ---------------------------------------------------------------------------
// Final reduction: mean over batch of (forward - gold).
// ---------------------------------------------------------------------------
__global__ __launch_bounds__(256) void crf_reduce_kernel(
    const float* __restrict__ fwd, const float* __restrict__ gold,
    float* __restrict__ out) {
    __shared__ float sh[256];
    int tid = threadIdx.x;
    float s = (fwd[tid] - gold[tid]) + (fwd[tid + 256] - gold[tid + 256]);
    sh[tid] = s;
    __syncthreads();
    for (int off = 128; off > 0; off >>= 1) {
        if (tid < off) sh[tid] += sh[tid + off];
        __syncthreads();
    }
    if (tid == 0) out[0] = sh[0] * (1.0f / (float)Bn);
}

extern "C" void kernel_launch(void* const* d_in, const int* in_sizes, int n_in,
                              void* d_out, int out_size, void* d_ws, size_t ws_size,
                              hipStream_t stream) {
    const float* feats = (const float*)d_in[0];
    const int*   tags  = (const int*)d_in[1];
    const float* trans = (const float*)d_in[2];
    float* out  = (float*)d_out;
    float* fwd  = (float*)d_ws;         // [512]
    float* gold = fwd + Bn;             // [512]

    crf_forward_kernel<<<Bn / 16, 256, 0, stream>>>(feats, trans, fwd);
    crf_gold_kernel<<<Bn / 256, 256, 0, stream>>>(feats, tags, trans, gold);
    crf_reduce_kernel<<<1, 256, 0, stream>>>(fwd, gold, out);
}
